// LLM_54073638257128
// MI455X (gfx1250) — matrix operand order repaired and checked by reading, not compiled, not run on hardware
//
#include <hip/hip_runtime.h>
#include <hip/hip_bf16.h>

// ---------------------------------------------------------------------------
// CDNA5 (gfx1250) transformer forward pass.
// Dominant cost: [4096x1024]x[1024x32000] vocab GEMM (~268 GFLOP) -> WMMA f16.
// GEMM: B tile (64N x 64K f16, 8KB) staged in LDS, double buffered, moved by
// the Tensor Data Mover (tensor_load_to_lds + s_wait_tensorcnt) issued by
// wave 0 while all 8 waves run WMMAs on the previous slab. All B fragments
// for a slab are loaded from LDS up front so the XDL pipe isn't gated on a
// per-WMMA s_wait_dscnt 0. Fallback path (no TDM builtin) does a cooperative
// global->LDS copy.
// Attention: flash-style, 1 wave per 16-query block, WMMA for QK^T and P@V.
// ---------------------------------------------------------------------------

typedef _Float16 v16h __attribute__((ext_vector_type(16)));
typedef _Float16 v8h  __attribute__((ext_vector_type(8)));
typedef float    v8f  __attribute__((ext_vector_type(8)));
typedef unsigned int u32x4 __attribute__((ext_vector_type(4)));
typedef int          i32x4 __attribute__((ext_vector_type(4)));
typedef int          i32x8 __attribute__((ext_vector_type(8)));

#if defined(__has_builtin)
#if __has_builtin(__builtin_amdgcn_tensor_load_to_lds) && \
    __has_builtin(__builtin_amdgcn_s_wait_tensorcnt)
#define USE_TDM 1
#endif
#endif

__device__ __forceinline__ v8f wmma_f16(v16h a, v16h b, v8f c) {
    // (neg_a, A, neg_b, B, c_mod, C, reuse_a, reuse_b)
    return __builtin_amdgcn_wmma_f32_16x16x32_f16(false, a, false, b, (short)0, c,
                                                  false, false);
}

__device__ __forceinline__ v16h cat8(v8h lo, v8h hi) {
    return __builtin_shufflevector(lo, hi, 0, 1, 2, 3, 4, 5, 6, 7,
                                           8, 9, 10, 11, 12, 13, 14, 15);
}

// A fragment (16x32 f16): lane m = lane%16, g = lane/16.
// elements 0..15 = A[m][k0 + g*16 .. +15] (contiguous 32B, matches B)
__device__ __forceinline__ v16h load_a_frag(const _Float16* __restrict__ rowptr,
                                            int k0, int g) {
    return *(const v16h*)(rowptr + k0 + g * 16);
}

// B fragment (32x16 f16) from a transposed [N][K] row:
// lane n = lane%16, elements 0..15 = row[k0 + g*16 .. +15] (contiguous 32B).
__device__ __forceinline__ v16h load_b_frag(const _Float16* __restrict__ rowptr,
                                            int k0, int g) {
    return *(const v16h*)(rowptr + k0 + g * 16);
}

// ---------------------------------------------------------------------------
// Embedding: h16[b*T+t][e] = f16(tok_emb[x[b,t]][e] + pos_emb[t][e])
// ---------------------------------------------------------------------------
__global__ __launch_bounds__(256) void embed_kernel(
    const int* __restrict__ x, const float* __restrict__ tok,
    const float* __restrict__ pos, _Float16* __restrict__ h16, int T, int E) {
    int row = blockIdx.x;          // b*T + t
    int t = row % T;
    const float* tp = tok + (size_t)x[row] * E;
    const float* pp = pos + (size_t)t * E;
    _Float16* op = h16 + (size_t)row * E;
    for (int e = threadIdx.x; e < E; e += blockDim.x)
        op[e] = (_Float16)(tp[e] + pp[e]);
}

// ---------------------------------------------------------------------------
// Pack QKV weights [H,E,HS] fp32 -> transposed f16 [N=H*HS][K=E]
// ---------------------------------------------------------------------------
__global__ __launch_bounds__(256) void pack_w_qkv(
    const float* __restrict__ Wq, const float* __restrict__ Wk,
    const float* __restrict__ Wv, _Float16* __restrict__ oq,
    _Float16* __restrict__ ok, _Float16* __restrict__ ov, int E, int HS) {
    const float* W = (blockIdx.y == 0) ? Wq : (blockIdx.y == 1) ? Wk : Wv;
    _Float16* O = (blockIdx.y == 0) ? oq : (blockIdx.y == 1) ? ok : ov;
    int idx = blockIdx.x * blockDim.x + threadIdx.x;  // over N*E
    int n = idx / E, e = idx % E;
    int h = n / HS, d = n % HS;
    O[(size_t)n * E + e] = (_Float16)W[((size_t)h * E + e) * HS + d];
}

// Generic transpose+cast: in [K][N] fp32 -> out [N][K] f16
__global__ __launch_bounds__(256) void transpose_pack_f16(
    const float* __restrict__ in, _Float16* __restrict__ out, int K, int N) {
    size_t idx = (size_t)blockIdx.x * blockDim.x + threadIdx.x;
    size_t n = idx / K, k = idx % K;
    out[n * K + k] = (_Float16)in[k * N + n];
}

// v [B,T,H,HS] f16 -> vT [B,H,HS,T] f16 (so P@V B-fragments are contiguous)
__global__ __launch_bounds__(256) void transpose_v_kernel(
    const _Float16* __restrict__ v, _Float16* __restrict__ vT, int T, int H,
    int HS) {
    size_t idx = (size_t)blockIdx.x * blockDim.x + threadIdx.x;
    int D = H * HS;
    int d = idx % HS;
    size_t r = idx / HS;
    int h = r % H;
    size_t r2 = r / H;
    int t = r2 % T;
    int b = (int)(r2 / T);
    vT[(((size_t)b * H + h) * HS + d) * T + t] =
        v[((size_t)b * T + t) * D + h * HS + d];
}

// ---------------------------------------------------------------------------
// GEMM: C[M,N] = A[M,K]f16 @ BT[N,K]f16 (+bias)(+relu), C fp32 or f16.
// Block = 256 threads = 8 waves; wave w -> rows [bx*128+w*16, +16), all waves
// share cols [by*64, +64). B slab (64 x 64 f16 = 8 KB) lives in LDS, double
// buffered; next slab is fetched by the TDM (wave 0 issues tensor_load_to_lds)
// while the current slab feeds 8 WMMAs/iteration.
// flags: bit0=relu, bit1=store f16.
// ---------------------------------------------------------------------------
#define KSTEP 64

__device__ __forceinline__ void tdm_issue_b_slab(const _Float16* gsrc,
                                                 unsigned lds_byte_addr, int K) {
#if defined(USE_TDM)
    unsigned long long ga = (unsigned long long)gsrc;
    u32x4 g0;
    g0[0] = 1u;                                   // count=1, user descriptor
    g0[1] = lds_byte_addr;                        // lds_addr (bytes)
    g0[2] = (unsigned)(ga & 0xFFFFFFFFull);       // global_addr[31:0]
    g0[3] = (unsigned)((ga >> 32) & 0x1FFFFFFull) // global_addr[56:32]
            | (2u << 30);                         // type = 2 ("image")
    i32x8 g1;
    g1[0] = 1 << 16;              // workgroup_mask=0, data_size=1 (2 bytes)
    g1[1] = (K & 0xFFFF) << 16;   // tensor_dim0[15:0] in bits 63:48
    g1[2] = (K >> 16) | (64 << 16);  // tensor_dim0[31:16] | tensor_dim1=64
    g1[3] = (KSTEP << 16);        // tile_dim0 = KSTEP (bits 127:112)
    g1[4] = 64;                   // tile_dim1 = 64 rows, tile_dim2 = 0
    g1[5] = K;                    // tensor_dim0_stride[31:0] = K
    g1[6] = 0;                    // stride0[47:32], stride1[15:0]
    g1[7] = 0;
    i32x4 z4 = {0, 0, 0, 0};
#if __clang_major__ >= 23
    i32x8 z8 = {0, 0, 0, 0, 0, 0, 0, 0};
    __builtin_amdgcn_tensor_load_to_lds(g0, g1, z4, z4, z8, 0);
#else
    __builtin_amdgcn_tensor_load_to_lds(g0, g1, z4, z4, 0);
#endif
#else
    (void)gsrc; (void)lds_byte_addr; (void)K;
#endif
}

__global__ __launch_bounds__(256) void gemm_f16_wmma(
    const _Float16* __restrict__ A, const _Float16* __restrict__ BT,
    const float* __restrict__ bias, void* __restrict__ Cout, int M, int N,
    int K, int flags) {
    const int wave = threadIdx.x >> 5;
    const int lane = threadIdx.x & 31;
    const int g = lane >> 4;
    const int ln = lane & 15;
    const int rowBase = blockIdx.x * 128 + wave * 16;
    const int colBase = blockIdx.y * 64;

    // double-buffered B slab: [buf][n(64)][k(KSTEP)]
    __shared__ __align__(128) _Float16 bsh[2][64 * KSTEP];

    const _Float16* Ar = A + (size_t)(rowBase + ln) * K;
    const _Float16* Bslab = BT + (size_t)colBase * K;  // 64 rows, stride K

#if !defined(USE_TDM)
    // fallback cooperative copy: thread t -> row t/4, 32B chunk t%4 (+ half)
    const int crow = threadIdx.x >> 2;
    const int cchk = threadIdx.x & 3;
#endif

    // ---- stage slab 0 into buffer 0 --------------------------------------
#if defined(USE_TDM)
    if (wave == 0)
        tdm_issue_b_slab(Bslab, (unsigned)(unsigned long long)&bsh[0][0], K);
#else
    {
        v16h t0 = *(const v16h*)(Bslab + (size_t)crow * K + cchk * 16);
        *(v16h*)&bsh[0][crow * KSTEP + cchk * 16] = t0;
        v16h t1 = *(const v16h*)(Bslab + (size_t)crow * K + 32 + cchk * 16);
        *(v16h*)&bsh[0][crow * KSTEP + 32 + cchk * 16] = t1;
    }
#endif

    v8f acc[4] = {};
    int cur = 0;
    for (int k0 = 0; k0 < K; k0 += KSTEP) {
        const int nxt = cur ^ 1;
        const bool have_next = (k0 + KSTEP) < K;

        // ---- kick off next slab while we compute on the current one ------
        if (have_next) {
#if defined(USE_TDM)
            if (wave == 0)
                tdm_issue_b_slab(Bslab + (k0 + KSTEP),
                                 (unsigned)(unsigned long long)&bsh[nxt][0], K);
#else
            v16h t0 = *(const v16h*)(Bslab + (size_t)crow * K + k0 + KSTEP +
                                     cchk * 16);
            *(v16h*)&bsh[nxt][crow * KSTEP + cchk * 16] = t0;
            v16h t1 = *(const v16h*)(Bslab + (size_t)crow * K + k0 + KSTEP + 32 +
                                     cchk * 16);
            *(v16h*)&bsh[nxt][crow * KSTEP + 32 + cchk * 16] = t1;
#endif
        }

#if defined(USE_TDM)
        // current slab complete when <= (outstanding next) transfers in flight.
        // s_wait_tensorcnt needs a constant immediate -> uniform branch.
        if (wave == 0) {
            if (have_next)
                __builtin_amdgcn_s_wait_tensorcnt(1);
            else
                __builtin_amdgcn_s_wait_tensorcnt(0);
        }
#endif
        __syncthreads();  // current slab visible to all waves

        // ---- compute: 2 A fragments x 4 N tiles = 8 WMMAs ----------------
        __builtin_prefetch(Ar + k0 + 2 * KSTEP, 0, 3);  // global_prefetch_b8
        v16h a0 = load_a_frag(Ar, k0, g);
        v16h a1 = load_a_frag(Ar, k0 + 32, g);
        const _Float16* bs = &bsh[cur][0];

        // load ALL B fragments first (one ds_load clause, draining waits)
        v16h bfr[8];
#pragma unroll
        for (int c = 0; c < 4; ++c) {
            const _Float16* brow = bs + (c * 16 + ln) * KSTEP;
            bfr[2 * c]     = load_b_frag(brow, 0, g);
            bfr[2 * c + 1] = load_b_frag(brow, 32, g);
        }
#pragma unroll
        for (int c = 0; c < 4; ++c) {
            acc[c] = wmma_f16(a0, bfr[2 * c], acc[c]);
            acc[c] = wmma_f16(a1, bfr[2 * c + 1], acc[c]);
        }
        __syncthreads();  // all readers done before this buffer is re-filled
        cur = nxt;
    }

#pragma unroll
    for (int c = 0; c < 4; ++c) {
#pragma unroll
        for (int i = 0; i < 8; ++i) {
            int row = rowBase + i + 8 * g;  // C/D layout: VGPR i -> M = i + 8g
            int col = colBase + c * 16 + ln;
            float v = acc[c][i];
            if (bias) v += bias[col];
            if (flags & 1) v = fmaxf(v, 0.0f);
            if (flags & 2)
                ((_Float16*)Cout)[(size_t)row * N + col] = (_Float16)v;
            else
                ((float*)Cout)[(size_t)row * N + col] = v;
        }
    }
}

// ---------------------------------------------------------------------------
// Flash attention, causal. One wave per (b, h, 16-query block).
// scores = (q @ k^T) * sqrt(HS); online softmax; acc = P @ V.
// P round-trips through LDS to convert C-layout -> A-layout (s_wait_dscnt).
// ---------------------------------------------------------------------------
__global__ __launch_bounds__(32) void attn_flash_wmma(
    const _Float16* __restrict__ q, const _Float16* __restrict__ kk,
    const _Float16* __restrict__ vT, _Float16* __restrict__ o, int T, int H,
    int HS) {
    const int lane = threadIdx.x & 31;
    const int g = lane >> 4;
    const int ln = lane & 15;
    const int qb = blockIdx.x;
    const int h = blockIdx.y;
    const int b = blockIdx.z;
    const int D = H * HS;
    const float NEG_INF = -__builtin_inff();

    __shared__ __align__(64) _Float16 pbuf[16 * 32];

    const _Float16* Qb = q + (size_t)b * T * D + h * HS;
    const _Float16* Kb = kk + (size_t)b * T * D + h * HS;
    const _Float16* Vb = vT + ((size_t)b * H + h) * HS * T;

    // Q fragments for this 16-row block (HS=64 -> two 16x32 A fragments)
    const _Float16* qr = Qb + (size_t)(qb * 16 + ln) * D;
    v16h aq0 = load_a_frag(qr, 0, g);
    v16h aq1 = load_a_frag(qr, 32, g);

    float Mrow[8], Lrow[8];
#pragma unroll
    for (int i = 0; i < 8; ++i) { Mrow[i] = NEG_INF; Lrow[i] = 0.0f; }
    v8f acc[4] = {};

    const float scale = 8.0f;  // reference multiplies by sqrt(HS) = 8

    for (int t0 = 0; t0 < (qb + 1) * 16; t0 += 32) {
        // ---- scores: load all 4 K fragments, then 4 WMMAs -----------------
        const _Float16* k0p = Kb + (size_t)(t0 + ln) * D;
        const _Float16* k1p = Kb + (size_t)(t0 + 16 + ln) * D;
        v16h bk00 = load_b_frag(k0p, 0, g);
        v16h bk01 = load_b_frag(k0p, 32, g);
        v16h bk10 = load_b_frag(k1p, 0, g);
        v16h bk11 = load_b_frag(k1p, 32, g);
        v8f S0 = {}, S1 = {};
        S0 = wmma_f16(aq0, bk00, S0);
        S0 = wmma_f16(aq1, bk01, S0);
        S1 = wmma_f16(aq0, bk10, S1);
        S1 = wmma_f16(aq1, bk11, S1);

        // ---- online softmax (row stats per lane-half via shfl_xor) --------
        float alpha_i[8];
#pragma unroll
        for (int i = 0; i < 8; ++i) {
            int qg = qb * 16 + i + 8 * g;  // this VGPR's query row
            float s0 = (t0 + ln <= qg) ? S0[i] * scale : NEG_INF;
            float s1 = (t0 + 16 + ln <= qg) ? S1[i] * scale : NEG_INF;
            float r = fmaxf(s0, s1);
#pragma unroll
            for (int off = 1; off < 16; off <<= 1)
                r = fmaxf(r, __shfl_xor(r, off, 32));
            float Mn = fmaxf(Mrow[i], r);
            float al = __expf(Mrow[i] - Mn);
            float p0 = __expf(s0 - Mn);
            float p1 = __expf(s1 - Mn);
            float rs = p0 + p1;
#pragma unroll
            for (int off = 1; off < 16; off <<= 1)
                rs += __shfl_xor(rs, off, 32);
            Lrow[i] = Lrow[i] * al + rs;
            Mrow[i] = Mn;
            alpha_i[i] = al;
            // C-layout -> LDS (row-major 16x32 P tile)
            pbuf[(i + 8 * g) * 32 + ln] = (_Float16)p0;
            pbuf[(i + 8 * g) * 32 + 16 + ln] = (_Float16)p1;
        }
#pragma unroll
        for (int c = 0; c < 4; ++c)
#pragma unroll
            for (int i = 0; i < 8; ++i) acc[c][i] *= alpha_i[i];

        asm volatile("s_wait_dscnt 0" ::: "memory");  // LDS stores visible

        // ---- P @ V: A = P (16x32 from LDS), B = vT rows (contiguous keys) -
        const _Float16* pr = pbuf + ln * 32;
        v16h ap = load_a_frag(pr, 0, g);
        v16h bv0 = *(const v16h*)(Vb + (size_t)(0 * 16 + ln) * T + t0 + g * 16);
        v16h bv1 = *(const v16h*)(Vb + (size_t)(1 * 16 + ln) * T + t0 + g * 16);
        v16h bv2 = *(const v16h*)(Vb + (size_t)(2 * 16 + ln) * T + t0 + g * 16);
        v16h bv3 = *(const v16h*)(Vb + (size_t)(3 * 16 + ln) * T + t0 + g * 16);
        acc[0] = wmma_f16(ap, bv0, acc[0]);
        acc[1] = wmma_f16(ap, bv1, acc[1]);
        acc[2] = wmma_f16(ap, bv2, acc[2]);
        acc[3] = wmma_f16(ap, bv3, acc[3]);
    }

    // ---- normalize and store [b, t, h, d] --------------------------------
#pragma unroll
    for (int c = 0; c < 4; ++c)
#pragma unroll
        for (int i = 0; i < 8; ++i) {
            int qg = qb * 16 + i + 8 * g;
            o[((size_t)b * T + qg) * D + h * HS + c * 16 + ln] =
                (_Float16)(acc[c][i] / Lrow[i]);
        }
}

// ---------------------------------------------------------------------------
// Loss: per-row log-softmax NLL (deterministic two-pass), then mean.
// ---------------------------------------------------------------------------
__global__ __launch_bounds__(256) void row_loss_kernel(
    const float* __restrict__ logits, const int* __restrict__ y,
    float* __restrict__ rl, int V) {
    __shared__ float red[256];
    const int row = blockIdx.x;
    const float* lp = logits + (size_t)row * V;
    float mx = -__builtin_inff();
    for (int j = threadIdx.x; j < V; j += 256) mx = fmaxf(mx, lp[j]);
    red[threadIdx.x] = mx;
    __syncthreads();
    for (int s = 128; s > 0; s >>= 1) {
        if (threadIdx.x < s)
            red[threadIdx.x] = fmaxf(red[threadIdx.x], red[threadIdx.x + s]);
        __syncthreads();
    }
    mx = red[0];
    __syncthreads();
    float sum = 0.0f;
    for (int j = threadIdx.x; j < V; j += 256) sum += __expf(lp[j] - mx);
    red[threadIdx.x] = sum;
    __syncthreads();
    for (int s = 128; s > 0; s >>= 1) {
        if (threadIdx.x < s) red[threadIdx.x] += red[threadIdx.x + s];
        __syncthreads();
    }
    if (threadIdx.x == 0)
        rl[row] = -(lp[y[row]] - mx - __logf(red[0]));
}

__global__ __launch_bounds__(256) void loss_finalize(
    const float* __restrict__ rl, float* __restrict__ out, int n) {
    __shared__ float red[256];
    float s = 0.0f;
    for (int j = threadIdx.x; j < n; j += 256) s += rl[j];
    red[threadIdx.x] = s;
    __syncthreads();
    for (int st = 128; st > 0; st >>= 1) {
        if (threadIdx.x < st) red[threadIdx.x] += red[threadIdx.x + st];
        __syncthreads();
    }
    if (threadIdx.x == 0) out[0] = red[0] / (float)n;
}

// ---------------------------------------------------------------------------
extern "C" void kernel_launch(void* const* d_in, const int* in_sizes, int n_in,
                              void* d_out, int out_size, void* d_ws,
                              size_t ws_size, hipStream_t stream) {
    (void)in_sizes; (void)n_in; (void)out_size; (void)ws_size;
    const int B = 4, T = 1024, E = 1024, H = 16, HS = 64, V = 32000;
    const int D = H * HS, BT = B * T;

    const int* x = (const int*)d_in[0];
    const int* y = (const int*)d_in[1];
    const float* tok = (const float*)d_in[2];
    const float* pos = (const float*)d_in[3];
    const float* Wq = (const float*)d_in[4];
    const float* Wk = (const float*)d_in[5];
    const float* Wv = (const float*)d_in[6];
    const float* ffw = (const float*)d_in[7];
    const float* ffb = (const float*)d_in[8];
    const float* fcw = (const float*)d_in[9];
    const float* fcb = (const float*)d_in[10];

    char* ws = (char*)d_ws;
    auto alloc = [&](size_t bytes) {
        char* p = ws;
        ws += (bytes + 255) & ~(size_t)255;
        return p;
    };
    _Float16* h16 = (_Float16*)alloc((size_t)BT * E * 2);
    _Float16* q16 = (_Float16*)alloc((size_t)BT * D * 2);
    _Float16* k16 = (_Float16*)alloc((size_t)BT * D * 2);
    _Float16* v16 = (_Float16*)alloc((size_t)BT * D * 2);
    _Float16* vT16 = (_Float16*)alloc((size_t)BT * D * 2);
    _Float16* o16 = (_Float16*)alloc((size_t)BT * D * 2);
    _Float16* ff16 = (_Float16*)alloc((size_t)BT * D * 2);
    _Float16* wqT = (_Float16*)alloc((size_t)D * E * 2);
    _Float16* wkT = (_Float16*)alloc((size_t)D * E * 2);
    _Float16* wvT = (_Float16*)alloc((size_t)D * E * 2);
    _Float16* ffT = (_Float16*)alloc((size_t)D * D * 2);
    _Float16* fcT = (_Float16*)alloc((size_t)V * D * 2);
    float* rl = (float*)alloc((size_t)BT * 4);

    float* logits = (float*)d_out;
    float* lossp = (float*)d_out + (size_t)BT * V;

    embed_kernel<<<BT, 256, 0, stream>>>(x, tok, pos, h16, T, E);
    pack_w_qkv<<<dim3((D * E) / 256, 3), 256, 0, stream>>>(Wq, Wk, Wv, wqT, wkT,
                                                           wvT, E, HS);
    transpose_pack_f16<<<(D * D) / 256, 256, 0, stream>>>(ffw, ffT, D, D);
    transpose_pack_f16<<<(unsigned)(((size_t)D * V) / 256), 256, 0, stream>>>(
        fcw, fcT, D, V);

    // QKV projections: [BT,E] @ [E,D] -> f16
    gemm_f16_wmma<<<dim3(BT / 128, D / 64), 256, 0, stream>>>(h16, wqT, nullptr,
                                                              q16, BT, D, E, 2);
    gemm_f16_wmma<<<dim3(BT / 128, D / 64), 256, 0, stream>>>(h16, wkT, nullptr,
                                                              k16, BT, D, E, 2);
    gemm_f16_wmma<<<dim3(BT / 128, D / 64), 256, 0, stream>>>(h16, wvT, nullptr,
                                                              v16, BT, D, E, 2);
    transpose_v_kernel<<<(unsigned)(((size_t)BT * D) / 256), 256, 0, stream>>>(
        v16, vT16, T, H, HS);

    attn_flash_wmma<<<dim3(T / 16, H, B), 32, 0, stream>>>(q16, k16, vT16, o16,
                                                           T, H, HS);

    // FF: relu(o @ ff_w + ff_b) -> f16
    gemm_f16_wmma<<<dim3(BT / 128, D / 64), 256, 0, stream>>>(o16, ffT, ffb,
                                                              ff16, BT, D, D, 3);
    // Logits: ff @ fc_w + fc_b -> fp32 directly into d_out
    gemm_f16_wmma<<<dim3(BT / 128, V / 64), 256, 0, stream>>>(
        ff16, fcT, fcb, logits, BT, V, D, 0);

    row_loss_kernel<<<BT, 256, 0, stream>>>(logits, y, rl, V);
    loss_finalize<<<1, 256, 0, stream>>>(rl, lossp, BT);
}
